// Tagger_10325101380107
// MI455X (gfx1250) — matrix operand order repaired and checked by reading, not compiled, not run on hardware
//
#include <hip/hip_runtime.h>
#include <hip/hip_bf16.h>

typedef __attribute__((ext_vector_type(16))) _Float16 v16h;
typedef __attribute__((ext_vector_type(8)))  float    v8f;

#define NROWS 16384

__device__ __forceinline__ float sigmoidf(float v) { return 1.0f / (1.0f + __expf(-v)); }

// ---------------------------------------------------------------------------
// Kernel 1: char-level LSTM, batch-parallel. One wave per 16 batch rows.
// gates(16x80) = [x(10)|h(20)|pad(2)](16x32,f16)  @  W^T(32x80,f16)  via
// 5x v_wmma_f32_16x16x32_f16 per step, 30 steps.
// Reference inits h0/c0 with JAX threefry xavier noise; approximated as 0 here.
// ---------------------------------------------------------------------------
__global__ __launch_bounds__(32) void char_lstm_kernel(
    const int*   __restrict__ x,        // [N][30]
    const float* __restrict__ emb,      // [128][10]
    const float* __restrict__ Wih,      // [80][10]
    const float* __restrict__ Whh,      // [80][20]
    const float* __restrict__ bih,      // [80]
    const float* __restrict__ bhh,      // [80]
    float*       __restrict__ word_emb) // [N][20]
{
    __shared__ _Float16 Wt[32][80];   // B staging: Wt[k][g] = W[g][k]
    __shared__ _Float16 At[16][32];   // A staging: [x | h | pad]
    __shared__ float    Gt[16][80];   // gate pre-activations
    __shared__ float    Cs[16][20];   // cell state
    __shared__ float    Bs[80];       // bias

    const int lane = threadIdx.x;
    const int r0   = blockIdx.x * 16;

    // Stage transposed combined weights (f16) + bias; zero h-part of A and C.
    for (int i = lane; i < 32 * 80; i += 32) {
        int k = i / 80, g = i % 80;
        float w = 0.0f;
        if (k < 10)      w = Wih[g * 10 + k];
        else if (k < 30) w = Whh[g * 20 + (k - 10)];
        Wt[k][g] = (_Float16)w;
    }
    for (int g = lane; g < 80; g += 32) Bs[g] = bih[g] + bhh[g];
    for (int i = lane; i < 16 * 20; i += 32) {
        int m = i / 20, j = i % 20;
        At[m][10 + j] = (_Float16)0.0f;
        Cs[m][j]      = 0.0f;
    }
    for (int m = lane; m < 16; m += 32) { At[m][30] = (_Float16)0.0f; At[m][31] = (_Float16)0.0f; }
    __syncthreads();

    // Persistent B fragments (5 gate tiles), per ISA 16-bit 32x16 B layout.
    const int l15   = lane & 15;
    const int kbB   = (lane < 16) ? 0 : 8;   // B fragment K base
    const int kbA   = (lane < 16) ? 0 : 8;   // A fragment K base (within 32-chunk)
    const int mbase = (lane < 16) ? 0 : 8;   // C/D row base
    v16h Bfrag[5];
    #pragma unroll
    for (int nt = 0; nt < 5; ++nt)
        #pragma unroll
        for (int i = 0; i < 8; ++i) {
            Bfrag[nt][i]     = Wt[kbB + i][nt * 16 + l15];
            Bfrag[nt][8 + i] = Wt[16 + kbB + i][nt * 16 + l15];
        }

    for (int t = 0; t < 30; ++t) {
        // Gather char embeddings for this timestep into A[:,0:10].
        for (int i = lane; i < 160; i += 32) {
            int m = i / 10, k = i % 10;
            int cid = x[(r0 + m) * 30 + t];
            At[m][k] = (_Float16)emb[cid * 10 + k];
        }
        __syncthreads();

        // A fragment (16-bit 16x32 layout).
        v16h Af;
        #pragma unroll
        for (int i = 0; i < 8; ++i) {
            Af[i]     = At[l15][kbA + i];
            Af[8 + i] = At[l15][16 + kbA + i];
        }

        // gates = A @ W^T + b  (5 WMMAs)
        #pragma unroll
        for (int nt = 0; nt < 5; ++nt) {
            float b = Bs[nt * 16 + l15];
            v8f acc;
            #pragma unroll
            for (int r = 0; r < 8; ++r) acc[r] = b;
            acc = __builtin_amdgcn_wmma_f32_16x16x32_f16(
                false, Af, false, Bfrag[nt], (short)0, acc, false, false);
            #pragma unroll
            for (int r = 0; r < 8; ++r) Gt[mbase + r][nt * 16 + l15] = acc[r];
        }
        __syncthreads();

        // Pointwise cell update: 320 (m,j) pairs, 10 per lane.
        #pragma unroll
        for (int q = 0; q < 10; ++q) {
            int idx = lane * 10 + q;
            int m = idx / 20, j = idx % 20;
            float gi = Gt[m][j], gf = Gt[m][20 + j], gg = Gt[m][40 + j], go = Gt[m][60 + j];
            float cn = sigmoidf(gf) * Cs[m][j] + sigmoidf(gi) * tanhf(gg);
            float hn = sigmoidf(go) * tanhf(cn);
            Cs[m][j]      = cn;
            At[m][10 + j] = (_Float16)hn;
        }
        __syncthreads();
    }

    for (int i = lane; i < 16 * 20; i += 32) {
        int m = i / 20, j = i % 20;
        word_emb[(r0 + m) * 20 + j] = (float)At[m][10 + j];
    }
}

// ---------------------------------------------------------------------------
// Kernel 2: word LSTM layer 1 (hid=30, in=20, 120 gate rows). Batch 1 ->
// latency chain: one block per direction, one thread per gate row, weights in
// LDS. block 0 = forward (f1), block 1 = reverse (b1). Output seq2[t][0:30]=f1,
// [30:60]=b1 (reference keeps b1 in scan order -- NOT re-reversed).
// ---------------------------------------------------------------------------
__global__ __launch_bounds__(128) void word_lstm1_kernel(
    const float* __restrict__ we, // [N][20]
    const float* __restrict__ fWih, const float* __restrict__ fWhh,
    const float* __restrict__ fbih, const float* __restrict__ fbhh,
    const float* __restrict__ bWih, const float* __restrict__ bWhh,
    const float* __restrict__ bbih, const float* __restrict__ bbhh,
    float* __restrict__ seq2)     // [N][60]
{
    __shared__ float W[120][52];  // [g][0:20]=Wih row, [20:50]=Whh row
    __shared__ float bias[120];
    __shared__ float h[30], c[30], g[120], xin[20];

    const bool rev = (blockIdx.x == 1);
    const float* Wihp = rev ? bWih : fWih;
    const float* Whhp = rev ? bWhh : fWhh;
    const float* bihp = rev ? bbih : fbih;
    const float* bhhp = rev ? bbhh : fbhh;
    const int tid = threadIdx.x;

    for (int i = tid; i < 120 * 20; i += 128) W[i / 20][i % 20]      = Wihp[i];
    for (int i = tid; i < 120 * 30; i += 128) W[i / 30][20 + i % 30] = Whhp[i];
    if (tid < 120) bias[tid] = bihp[tid] + bhhp[tid];
    if (tid < 30) { h[tid] = 0.0f; c[tid] = 0.0f; } // ref: threefry xavier; approx 0
    __syncthreads();

    for (int t = 0; t < NROWS; ++t) {
        const int src = rev ? (NROWS - 1 - t) : t;
        if (tid < 20) xin[tid] = we[src * 20 + tid];
        if (tid == 0) {
            int nxt = rev ? src - 4 : src + 4;
            if (nxt >= 0 && nxt < NROWS) __builtin_prefetch(&we[nxt * 20], 0, 1);
        }
        __syncthreads();
        if (tid < 120) {
            float a = bias[tid];
            #pragma unroll 4
            for (int k = 0; k < 20; ++k) a += W[tid][k] * xin[k];
            #pragma unroll 4
            for (int k = 0; k < 30; ++k) a += W[tid][20 + k] * h[k];
            g[tid] = a;
        }
        __syncthreads();
        if (tid < 30) {
            float cn = sigmoidf(g[30 + tid]) * c[tid] + sigmoidf(g[tid]) * tanhf(g[60 + tid]);
            float hn = sigmoidf(g[90 + tid]) * tanhf(cn);
            c[tid] = cn; h[tid] = hn;
            seq2[t * 60 + (rev ? 30 : 0) + tid] = hn;
        }
        __syncthreads();
    }
}

// ---------------------------------------------------------------------------
// Kernel 3: word LSTM layer 2 (hid=50, in=60, 200 gate rows). Same structure;
// weights stored f16 in LDS (200x112) to stay within static LDS budget.
// ---------------------------------------------------------------------------
__global__ __launch_bounds__(256) void word_lstm2_kernel(
    const float* __restrict__ seq2, // [N][60]
    const float* __restrict__ fWih, const float* __restrict__ fWhh,
    const float* __restrict__ fbih, const float* __restrict__ fbhh,
    const float* __restrict__ bWih, const float* __restrict__ bWhh,
    const float* __restrict__ bbih, const float* __restrict__ bbhh,
    float* __restrict__ layer2)     // [N][100]
{
    __shared__ _Float16 W[200][112]; // [g][0:60]=Wih, [60:110]=Whh
    __shared__ float bias[200];
    __shared__ float h[50], c[50], g[200], xin[60];

    const bool rev = (blockIdx.x == 1);
    const float* Wihp = rev ? bWih : fWih;
    const float* Whhp = rev ? bWhh : fWhh;
    const float* bihp = rev ? bbih : fbih;
    const float* bhhp = rev ? bbhh : fbhh;
    const int tid = threadIdx.x;

    for (int i = tid; i < 200 * 60; i += 256) W[i / 60][i % 60]      = (_Float16)Wihp[i];
    for (int i = tid; i < 200 * 50; i += 256) W[i / 50][60 + i % 50] = (_Float16)Whhp[i];
    if (tid < 200) bias[tid] = bihp[tid] + bhhp[tid];
    if (tid < 50) { h[tid] = 0.0f; c[tid] = 0.0f; } // ref: threefry xavier; approx 0
    __syncthreads();

    for (int t = 0; t < NROWS; ++t) {
        const int src = rev ? (NROWS - 1 - t) : t;
        if (tid < 60) xin[tid] = seq2[src * 60 + tid];
        if (tid == 0) {
            int nxt = rev ? src - 4 : src + 4;
            if (nxt >= 0 && nxt < NROWS) __builtin_prefetch(&seq2[nxt * 60], 0, 1);
        }
        __syncthreads();
        if (tid < 200) {
            float a = bias[tid];
            #pragma unroll 4
            for (int k = 0; k < 60; ++k) a += (float)W[tid][k] * xin[k];
            #pragma unroll 4
            for (int k = 0; k < 50; ++k) a += (float)W[tid][60 + k] * h[k];
            g[tid] = a;
        }
        __syncthreads();
        if (tid < 50) {
            float cn = sigmoidf(g[50 + tid]) * c[tid] + sigmoidf(g[tid]) * tanhf(g[100 + tid]);
            float hn = sigmoidf(g[150 + tid]) * tanhf(cn);
            c[tid] = cn; h[tid] = hn;
            layer2[t * 100 + (rev ? 50 : 0) + tid] = hn;
        }
        __syncthreads();
    }
}

// ---------------------------------------------------------------------------
// Kernel 4: output head. out = tanh(layer2 @ l1W^T + l1b) @ l2W^T + l2b.
// 4 waves/block, 16 rows/wave. GEMM1: K=100->4x32, N=100->7 tiles (28 WMMAs).
// GEMM2: K=100->4x32, N=17->2 tiles (8 WMMAs). A-buffer reused for tanh(H).
// ---------------------------------------------------------------------------
__global__ __launch_bounds__(128) void head_kernel(
    const float* __restrict__ layer2, // [N][100]
    const float* __restrict__ l1W,    // [100][100]
    const float* __restrict__ l1b,    // [100]
    const float* __restrict__ l2W,    // [17][100]
    const float* __restrict__ l2b,    // [17]
    float* __restrict__ out)          // [N][17]
{
    __shared__ _Float16 W1t[128][112];      // W1t[k][n] = l1W[n][k], zero-padded
    __shared__ _Float16 W2t[128][32];       // W2t[k][n] = l2W[n][k], zero-padded
    __shared__ _Float16 Abuf[4][16][128];   // per-wave activations / hidden

    const int tid  = threadIdx.x;
    const int lane = tid & 31;
    const int wave = tid >> 5;

    for (int i = tid; i < 128 * 112; i += 128) {
        int k = i / 112, n = i % 112;
        W1t[k][n] = (_Float16)((k < 100 && n < 100) ? l1W[n * 100 + k] : 0.0f);
    }
    for (int i = tid; i < 128 * 32; i += 128) {
        int k = i / 32, n = i % 32;
        W2t[k][n] = (_Float16)((k < 100 && n < 17) ? l2W[n * 100 + k] : 0.0f);
    }
    __syncthreads();

    const int r0 = blockIdx.x * 64 + wave * 16;

    // Stage this wave's 16x100 input tile (pad K to 128).
    for (int i = lane; i < 16 * 128; i += 32) {
        int m = i >> 7, k = i & 127;
        Abuf[wave][m][k] = (_Float16)((k < 100) ? layer2[(r0 + m) * 100 + k] : 0.0f);
    }
    __syncthreads();

    const int l15   = lane & 15;
    const int kbA   = (lane < 16) ? 0 : 8;
    const int kbB   = (lane < 16) ? 0 : 8;
    const int mbase = (lane < 16) ? 0 : 8;

    // GEMM1: hidden = A @ W1^T + b1, accumulate all 7 N-tiles before tanh.
    v8f accs[7];
    #pragma unroll
    for (int nt = 0; nt < 7; ++nt) {
        int col = nt * 16 + l15;
        float b = (col < 100) ? l1b[col] : 0.0f;
        #pragma unroll
        for (int r = 0; r < 8; ++r) accs[nt][r] = b;
        #pragma unroll
        for (int kc = 0; kc < 4; ++kc) {
            v16h a, bf;
            #pragma unroll
            for (int i = 0; i < 8; ++i) {
                a[i]     = Abuf[wave][l15][kc * 32 + kbA + i];
                a[8 + i] = Abuf[wave][l15][kc * 32 + 16 + kbA + i];
            }
            #pragma unroll
            for (int i = 0; i < 8; ++i) {
                bf[i]     = W1t[kc * 32 + kbB + i][nt * 16 + l15];
                bf[8 + i] = W1t[kc * 32 + 16 + kbB + i][nt * 16 + l15];
            }
            accs[nt] = __builtin_amdgcn_wmma_f32_16x16x32_f16(
                false, a, false, bf, (short)0, accs[nt], false, false);
        }
    }
    __syncthreads();
    // Overwrite Abuf with tanh(hidden); zero pad cols 112..127.
    #pragma unroll
    for (int nt = 0; nt < 7; ++nt)
        #pragma unroll
        for (int r = 0; r < 8; ++r)
            Abuf[wave][mbase + r][nt * 16 + l15] = (_Float16)tanhf(accs[nt][r]);
    for (int i = lane; i < 16 * 16; i += 32)
        Abuf[wave][i >> 4][112 + (i & 15)] = (_Float16)0.0f;
    __syncthreads();

    // GEMM2: out = H @ W2^T + b2 (N padded to 32, store cols < 17).
    #pragma unroll
    for (int nt = 0; nt < 2; ++nt) {
        int col = nt * 16 + l15;
        float b = (col < 17) ? l2b[col] : 0.0f;
        v8f acc;
        #pragma unroll
        for (int r = 0; r < 8; ++r) acc[r] = b;
        #pragma unroll
        for (int kc = 0; kc < 4; ++kc) {
            v16h a, bf;
            #pragma unroll
            for (int i = 0; i < 8; ++i) {
                a[i]     = Abuf[wave][l15][kc * 32 + kbA + i];
                a[8 + i] = Abuf[wave][l15][kc * 32 + 16 + kbA + i];
            }
            #pragma unroll
            for (int i = 0; i < 8; ++i) {
                bf[i]     = W2t[kc * 32 + kbB + i][nt * 16 + l15];
                bf[8 + i] = W2t[kc * 32 + 16 + kbB + i][nt * 16 + l15];
            }
            acc = __builtin_amdgcn_wmma_f32_16x16x32_f16(
                false, a, false, bf, (short)0, acc, false, false);
        }
        if (col < 17) {
            #pragma unroll
            for (int r = 0; r < 8; ++r)
                out[(r0 + mbase + r) * 17 + col] = acc[r];
        }
    }
}

// ---------------------------------------------------------------------------
extern "C" void kernel_launch(void* const* d_in, const int* in_sizes, int n_in,
                              void* d_out, int out_size, void* d_ws, size_t ws_size,
                              hipStream_t stream) {
    const int*   x        = (const int*)  d_in[0];
    const float* char_emb = (const float*)d_in[1];
    const float* c_Wih    = (const float*)d_in[2];
    const float* c_Whh    = (const float*)d_in[3];
    const float* c_bih    = (const float*)d_in[4];
    const float* c_bhh    = (const float*)d_in[5];
    const float* f1_Wih   = (const float*)d_in[6];
    const float* f1_Whh   = (const float*)d_in[7];
    const float* f1_bih   = (const float*)d_in[8];
    const float* f1_bhh   = (const float*)d_in[9];
    const float* b1_Wih   = (const float*)d_in[10];
    const float* b1_Whh   = (const float*)d_in[11];
    const float* b1_bih   = (const float*)d_in[12];
    const float* b1_bhh   = (const float*)d_in[13];
    const float* f2_Wih   = (const float*)d_in[14];
    const float* f2_Whh   = (const float*)d_in[15];
    const float* f2_bih   = (const float*)d_in[16];
    const float* f2_bhh   = (const float*)d_in[17];
    const float* b2_Wih   = (const float*)d_in[18];
    const float* b2_Whh   = (const float*)d_in[19];
    const float* b2_bih   = (const float*)d_in[20];
    const float* b2_bhh   = (const float*)d_in[21];
    const float* l1_W     = (const float*)d_in[22];
    const float* l1_b     = (const float*)d_in[23];
    const float* l2_W     = (const float*)d_in[24];
    const float* l2_b     = (const float*)d_in[25];
    float* out = (float*)d_out;

    char* ws = (char*)d_ws;
    float* word_emb = (float*)(ws);                                   // N*20
    float* seq2     = (float*)(ws + (size_t)NROWS * 20 * 4);          // N*60
    float* layer2   = (float*)(ws + (size_t)NROWS * 20 * 4
                                  + (size_t)NROWS * 60 * 4);          // N*100

    char_lstm_kernel<<<NROWS / 16, 32, 0, stream>>>(
        x, char_emb, c_Wih, c_Whh, c_bih, c_bhh, word_emb);

    word_lstm1_kernel<<<2, 128, 0, stream>>>(
        word_emb, f1_Wih, f1_Whh, f1_bih, f1_bhh,
                  b1_Wih, b1_Whh, b1_bih, b1_bhh, seq2);

    word_lstm2_kernel<<<2, 256, 0, stream>>>(
        seq2, f2_Wih, f2_Whh, f2_bih, f2_bhh,
              b2_Wih, b2_Whh, b2_bih, b2_bhh, layer2);

    head_kernel<<<NROWS / 64, 128, 0, stream>>>(
        layer2, l1_W, l1_b, l2_W, l2_b, out);
}
